// EnhancedMultiModalBiCameralCRSN_24902220382622
// MI455X (gfx1250) — matrix operand order repaired and checked by reading, not compiled, not run on hardware
//
#include <hip/hip_runtime.h>
#include <math.h>
#include <stdint.h>

// Problem constants (from reference)
#define EPSF  1e-6f
#define Bz    2048
#define Dz    256
#define DF    512      // 2*D (zf space)
#define ST    48       // stack slots
#define DEPTH 12
#define MT    16       // batch rows per workgroup (one WMMA M-tile)
#define TPB   512      // 16 waves; wave w <-> N-tile w (GEMM) or row w (elementwise)

typedef __attribute__((ext_vector_type(16))) __bf16 v16bf;
typedef __attribute__((ext_vector_type(8)))  __bf16 v8bf;
typedef __attribute__((ext_vector_type(8)))  float  v8f;
typedef __attribute__((ext_vector_type(8)))  int    v8i;
typedef __attribute__((ext_vector_type(4)))  float  v4f;

// Explicit global-address-space pointers: after laundering the weight base
// through an integer (to block LICM), we must re-assert addrspace(1) so the
// loads lower to global_load_b128 (LOADcnt only, SADDR+voffset addressing)
// instead of flat_load_b128 (LOADcnt+DScnt, 64-bit VGPR addressing).
typedef __attribute__((address_space(1))) const __bf16 gbf;
typedef __attribute__((address_space(1))) const v8bf   gv8bf;

__device__ __forceinline__ float wave_sum(float v) {
#pragma unroll
  for (int s = 16; s >= 1; s >>= 1) v += __shfl_xor(v, s, 32);
  return v;  // butterfly: all 32 lanes hold the sum
}

__device__ __forceinline__ v16bf cat8(v8bf lo, v8bf hi) {
  return __builtin_shufflevector(lo, hi, 0,1,2,3,4,5,6,7,8,9,10,11,12,13,14,15);
}

__device__ __forceinline__ v8f v8f_zero() {
  v8f z = {0.f,0.f,0.f,0.f,0.f,0.f,0.f,0.f};
  return z;
}

// A fragment: 16x32 bf16 tile from row-major [16][256] LDS source (ISA 7.12.2).
// lane<16: elems 0..7 <- K k0..k0+7, elems 8..15 <- K k0+16..k0+23 (k0 = kt*32)
// lane>=16: same with +8 K offset.  Two 16B LDS loads per lane.
__device__ __forceinline__ v16bf load_a_frag(const __bf16* base, int lane, int kt) {
  const __bf16* p = base + (lane & 15) * Dz + kt * 32 + ((lane >> 4) << 3);
  v8bf lo = *(const v8bf*)p;
  v8bf hi = *(const v8bf*)(p + 16);
  return cat8(lo, hi);
}

// B fragment: contraction over k of W[n][k] (row-major [256][256]); out = z @ W.T.
// B layout: n = lane&15, lanes<16 hold K=k0..k0+7,k0+16..k0+23; lanes>=16 same with +8 K offset.
__device__ __forceinline__ v16bf load_b_frag(gbf* W, int n0, int lane, int kt) {
  gbf* p = W + (size_t)(n0 + (lane & 15)) * Dz + kt * 32 + ((lane >> 4) << 3);
  v8bf lo = *(gv8bf*)p;
  v8bf hi = *(gv8bf*)(p + 16);
  return cat8(lo, hi);
}

// bf16 WMMA has no A/B NEG modifier (ISA: NEG = CNeg only) -> negate via sign bits.
__device__ __forceinline__ v16bf negf(v16bf a) {
  v8i u;
  __builtin_memcpy(&u, &a, sizeof(u));
  u ^= (int)0x80008000;
  v16bf r;
  __builtin_memcpy(&r, &u, sizeof(r));
  return r;
}

__device__ __forceinline__ v8f wmma_bf16(v16bf a, v16bf b, v8f c) {
  return __builtin_amdgcn_wmma_f32_16x16x32_bf16(false, a, false, b, (short)0, c,
                                                 false, false);
}

// (cr + i*ci) = (Ar + i*Ai) @ (Wr + i*Wi)^T  over K = 256
__device__ __forceinline__ void gemm_cplx(const __bf16* Ar, const __bf16* Ai,
                                          gbf* Wr, gbf* Wi,
                                          int n0, int lane, v8f& cr, v8f& ci) {
  v8f accr = v8f_zero();
  v8f acci = v8f_zero();
#pragma unroll
  for (int kt = 0; kt < 8; ++kt) {
    v16bf ar = load_a_frag(Ar, lane, kt);
    v16bf ai = load_a_frag(Ai, lane, kt);
    v16bf br = load_b_frag(Wr, n0, lane, kt);
    v16bf bi = load_b_frag(Wi, n0, lane, kt);
    accr = wmma_bf16(ar, br, accr);
    accr = wmma_bf16(negf(ai), bi, accr);   // - zi @ Wi.T
    acci = wmma_bf16(ai, br, acci);
    acci = wmma_bf16(ar, bi, acci);
  }
  cr = accr;
  ci = acci;
}

__global__ __launch_bounds__(256)
void convert_weights_kernel(const float* a, const float* b, const float* c,
                            const float* d, const float* e, const float* f,
                            __bf16* o) {
  int i = blockIdx.x * 256 + threadIdx.x;   // 256 blocks -> 65536 = 256*256
  o[i]             = (__bf16)a[i];
  o[i +     65536] = (__bf16)b[i];
  o[i + 2 * 65536] = (__bf16)c[i];
  o[i + 3 * 65536] = (__bf16)d[i];
  o[i + 4 * 65536] = (__bf16)e[i];
  o[i + 5 * 65536] = (__bf16)f[i];
}

__global__ __launch_bounds__(TPB)
void crsn_kernel(const float* __restrict__ z_real, const float* __restrict__ z_imag,
                 const float* __restrict__ ptr_in,
                 const float* __restrict__ ln_scale, const float* __restrict__ ln_shift,
                 const float* __restrict__ mod_bias,
                 const float* __restrict__ w_halt, const float* __restrict__ b_halt,
                 const float* __restrict__ w_ctrl, const float* __restrict__ b_ctrl,
                 const __bf16* __restrict__ wbf,   // 6 x [256][256] bf16 weights
                 float* __restrict__ hist,         // [B][DEPTH][DF] zf history
                 float* __restrict__ out)          // [B][DF]
{
  __shared__ __bf16 Ar[MT * Dz];    // bf16 activation tile (real)
  __shared__ __bf16 Ai[MT * Dz];    // bf16 activation tile (imag)
  __shared__ float  SF[MT * DF];    // f32 scratch: lin output / zf

  const int tid  = threadIdx.x;
  const int lane = tid & 31;
  const int wv   = tid >> 5;        // wave id
  const int r0   = blockIdx.x * MT; // global row base
  const int n0   = wv * 16;         // N-tile base (GEMM phases)
  const int m    = wv;              // row (elementwise phases)
  const int cb   = lane * 16;       // zf column base for phase E.3

  // initial z into LDS A-buffers
  for (int i = tid; i < MT * Dz; i += TPB) {
    int row = i >> 8, col = i & 255;
    Ar[i] = (__bf16)z_real[(size_t)(r0 + row) * Dz + col];
    Ai[i] = (__bf16)z_imag[(size_t)(r0 + row) * Dz + col];
  }

  // per-row stack state (wave m <-> row m): lane -> slot lane, lane<16 -> slot 32+lane
  float ptr_a = ptr_in[(size_t)(r0 + m) * ST + lane];
  float ptr_b = (lane < 16) ? ptr_in[(size_t)(r0 + m) * ST + 32 + lane] : 0.f;
  float coef_a[DEPTH], coef_b[DEPTH];   // memory == sum_tau coef[s][tau] * zf_tau
#pragma unroll
  for (int q = 0; q < DEPTH; ++q) { coef_a[q] = 0.f; coef_b[q] = 0.f; }
  float accv[16];                       // halting accumulator (ar||ai), 16 cols/lane
#pragma unroll
  for (int j = 0; j < 16; ++j) accv[j] = 0.f;
  float rem = 1.f;

  // zf-history region for THIS thread only (written and read by the same thread)
  float* hrow = hist + (size_t)(r0 + m) * DEPTH * DF + cb;

  __syncthreads();

  for (int t = 0; t < DEPTH; ++t) {
    // Launder the weight base pointer so the compiler cannot hoist the (loop-
    // invariant) weight-fragment loads out of the t-loop: hoisting spilled
    // ~384 VGPRs of fragments to scratch.  The addrspace(1) cast below keeps
    // the loads on the global (not flat) path.
    uint64_t wb_ = (uint64_t)wbf;
    asm volatile("" : "+s"(wb_));
    gbf* Wlr = (gbf*)wb_;
    gbf* Wli = Wlr + 65536;
    gbf* Wvr = Wlr + 2 * 65536;
    gbf* Wvi = Wlr + 3 * 65536;
    gbf* Wor = Wlr + 4 * 65536;
    gbf* Woi = Wlr + 5 * 65536;

    // ---------- Phase A: complex linear (w_lin) ----------
    v8f cr, ci;
    gemm_cplx(Ar, Ai, Wlr, Wli, n0, lane, cr, ci);
    {
      int n  = n0 + (lane & 15);
      int mb = (lane >> 4) << 3;
#pragma unroll
      for (int v = 0; v < 8; ++v) {
        SF[(mb + v) * DF + n]      = cr[v];
        SF[(mb + v) * DF + Dz + n] = ci[v];
      }
    }
    __syncthreads();

    // ---------- Phase B: magnitude layernorm + phase + modReLU ----------
    {
      float zr[8], zi[8], mg[8];
      float ssum = 0.f, ssq = 0.f;
#pragma unroll
      for (int j = 0; j < 8; ++j) {
        int c = lane * 8 + j;
        zr[j] = SF[m * DF + c];
        zi[j] = SF[m * DF + Dz + c];
        float r2 = sqrtf(zr[j] * zr[j] + zi[j] * zi[j]);
        mg[j] = r2 + EPSF;
        ssum += mg[j];
        ssq  += mg[j] * mg[j];
      }
      ssum = wave_sum(ssum);
      ssq  = wave_sum(ssq);
      float mean = ssum * (1.f / 256.f);
      float var  = (ssq - ssum * mean) * (1.f / 255.f);   // ddof=1
      float istd = rsqrtf(var + EPSF);
#pragma unroll
      for (int j = 0; j < 8; ++j) {
        int c = lane * 8 + j;
        float nm = (mg[j] - mean) * istd * ln_scale[c] + ln_shift[c];
        float r2 = mg[j] - EPSF;                          // |z|
        float cs, sn;
        if (r2 > 0.f) { float ir = 1.f / r2; cs = zr[j] * ir; sn = zi[j] * ir; }
        else          { cs = 1.f; sn = 0.f; }             // atan2(0,0)=0
        float nrm = fabsf(nm) + EPSF;                     // |nm*(cos,sin)| = |nm|
        float sc  = fmaxf(nrm + mod_bias[c], 0.f) / nrm;
        float s2  = nm * sc;
        Ar[m * Dz + c] = (__bf16)(s2 * cs);
        Ai[m * Dz + c] = (__bf16)(s2 * sn);
      }
    }
    __syncthreads();

    // ---------- Phase C: V projection (seq-len==1 => attention == identity) ----------
    v8f vr, vi;
    gemm_cplx(Ar, Ai, Wvr, Wvi, n0, lane, vr, vi);
    __syncthreads();                       // all waves done reading A
    {
      int n  = n0 + (lane & 15);
      int mb = (lane >> 4) << 3;
#pragma unroll
      for (int v = 0; v < 8; ++v) {
        Ar[(mb + v) * Dz + n] = (__bf16)vr[v];
        Ai[(mb + v) * Dz + n] = (__bf16)vi[v];
      }
    }
    __syncthreads();

    // ---------- Phase D: output projection -> pr,pi (to LDS only) ----------
    gemm_cplx(Ar, Ai, Wor, Woi, n0, lane, cr, ci);
    {
      int n  = n0 + (lane & 15);
      int mb = (lane >> 4) << 3;
#pragma unroll
      for (int v = 0; v < 8; ++v) {
        SF[(mb + v) * DF + n]      = cr[v];
        SF[(mb + v) * DF + Dz + n] = ci[v];
      }
    }
    __syncthreads();

    // ---------- Phase E: halt/ctrl + differentiable stack (wave m <-> row m) ----------
    // E.1 halt & ctrl GEMVs over zf[512]
    float hh = 0.f, c0 = 0.f, c1 = 0.f, c2 = 0.f;
#pragma unroll
    for (int i = 0; i < 16; ++i) {
      int c = lane + i * 32;
      float z = SF[m * DF + c];
      hh += z * w_halt[c];
      c0 += z * w_ctrl[c];
      c1 += z * w_ctrl[DF + c];
      c2 += z * w_ctrl[2 * DF + c];
    }
    hh = wave_sum(hh); c0 = wave_sum(c0); c1 = wave_sum(c1); c2 = wave_sum(c2);
    float halt = 1.f / (1.f + expf(-(hh + b_halt[0])));
    c0 += b_ctrl[0]; c1 += b_ctrl[1]; c2 += b_ctrl[2];
    float mx = fmaxf(c0, fmaxf(c1, c2));
    float e0 = expf(c0 - mx), e1 = expf(c1 - mx), e2 = expf(c2 - mx);
    float ei = 1.f / (e0 + e1 + e2);
    float push = e0 * ei, pop = e1 * ei, noop = e2 * ei;

    // E.2 pointer / coefficient dynamics (48 slots, wave-local via shuffles)
    float sAp  = __shfl(ptr_a, (lane - 1) & 31, 32);
    float sAn  = __shfl(ptr_a, (lane + 1) & 31, 32);
    float sBp  = __shfl(ptr_b, (lane - 1) & 31, 32);
    float sBn  = __shfl(ptr_b, (lane + 1) & 31, 32);
    float sA0  = __shfl(ptr_a, 0, 32);
    float sA31 = __shfl(ptr_a, 31, 32);
    float sB0  = __shfl(ptr_b, 0, 32);
    float sB15 = __shfl(ptr_b, 15, 32);
    float up_a   = (lane == 0)  ? sB15 : sAp;   // ptr[s-1 mod 48]
    float down_a = (lane == 31) ? sB0  : sAn;   // ptr[s+1 mod 48]
    float up_b   = (lane == 0)  ? sA31 : sBp;
    float down_b = (lane == 15) ? sA0  : sBn;
    bool  bval   = (lane < 16);
    float mix_a = push * up_a + pop * down_a + noop * ptr_a;
    float mix_b = bval ? (push * up_b + pop * down_b + noop * ptr_b) : 0.f;
    float qa = mix_a * mix_a, qb = mix_b * mix_b;
    float qs = wave_sum(qa + qb);
    float qiv = 1.f / (qs + EPSF);
    float np_a = qa * qiv, np_b = qb * qiv;
    float wm_a = push * up_a;
    float wm_b = bval ? push * up_b : 0.f;
    float oma = 1.f - wm_a, omb = 1.f - wm_b;
    float rho[DEPTH];
    float rho_cur = 0.f;
#pragma unroll
    for (int tau = 0; tau < DEPTH; ++tau) {
      if (tau < t)       { coef_a[tau] *= oma;  coef_b[tau] *= omb; }
      else if (tau == t) { coef_a[tau]  = wm_a; coef_b[tau]  = wm_b; }
      float pr = (tau <= t) ? (np_a * coef_a[tau] + np_b * coef_b[tau]) : 0.f;
      rho[tau] = (tau <= t) ? wave_sum(pr) : 0.f;
      if (tau == t) rho_cur = rho[tau];
    }
    ptr_a = np_a;
    ptr_b = np_b;

    // E.3 stack read (history FMA), history write, halting accumulation, carry z.
    // hist[(r0+m)*DEPTH + t][cb..cb+16) is written and later read ONLY by this
    // thread: same-wave store->load ordering is guaranteed, no fence needed,
    // and both directions are 16B-vector accesses.
    float zt[16], rr[16];
#pragma unroll
    for (int j = 0; j < 16; ++j) {
      zt[j] = SF[m * DF + cb + j];
      rr[j] = rho_cur * zt[j];              // tau==t term: hist[t] == zf_t
    }
    {
      v4f* hw = (v4f*)(hrow + (size_t)t * DF);
#pragma unroll
      for (int q = 0; q < 4; ++q) {
        v4f hv = { zt[q*4+0], zt[q*4+1], zt[q*4+2], zt[q*4+3] };
        hw[q] = hv;
      }
    }
#pragma unroll
    for (int tau = 0; tau < DEPTH; ++tau) {
      if (tau < t) {
        const v4f* hp = (const v4f*)(hrow + (size_t)tau * DF);
        float rt = rho[tau];
#pragma unroll
        for (int q = 0; q < 4; ++q) {
          v4f hv = hp[q];
          rr[q * 4 + 0] += rt * hv[0];
          rr[q * 4 + 1] += rt * hv[1];
          rr[q * 4 + 2] += rt * hv[2];
          rr[q * 4 + 3] += rt * hv[3];
        }
      }
    }
    float pw = halt * rem;
    rem = rem * (1.f - halt);
#pragma unroll
    for (int j = 0; j < 16; ++j) {
      float zn = zt[j] + rr[j];             // next carry z = pr + rr
      accv[j] += pw * zt[j];                // a += halt*rem * pr
      if (t == DEPTH - 1) {
        accv[j] += rem * zn;                // final: a += rem * z_carry
      } else {
        if (lane < 16) Ar[m * Dz + cb + j]        = (__bf16)zn;
        else           Ai[m * Dz + (cb - Dz) + j] = (__bf16)zn;
      }
    }
    __syncthreads();
  }

  // out = concat([ar, ai], -1): [B][512]
#pragma unroll
  for (int j = 0; j < 16; ++j)
    out[(size_t)(r0 + m) * DF + cb + j] = accv[j];
}

extern "C" void kernel_launch(void* const* d_in, const int* in_sizes, int n_in,
                              void* d_out, int out_size, void* d_ws, size_t ws_size,
                              hipStream_t stream) {
  const float* z_real   = (const float*)d_in[0];
  const float* z_imag   = (const float*)d_in[1];
  // d_in[2] = memory: all-zero initial state, represented implicitly (coef == 0)
  const float* ptr_in   = (const float*)d_in[3];
  const float* w_lin_r  = (const float*)d_in[4];
  const float* w_lin_i  = (const float*)d_in[5];
  const float* ln_scale = (const float*)d_in[6];
  const float* ln_shift = (const float*)d_in[7];
  const float* mod_bias = (const float*)d_in[8];
  // d_in[9..12] = wq/wk: dead (softmax over a length-1 sequence is identity)
  const float* wv_r     = (const float*)d_in[13];
  const float* wv_i     = (const float*)d_in[14];
  const float* wo_r     = (const float*)d_in[15];
  const float* wo_i     = (const float*)d_in[16];
  const float* w_halt   = (const float*)d_in[17];
  const float* b_halt   = (const float*)d_in[18];
  const float* w_ctrl   = (const float*)d_in[19];
  const float* b_ctrl   = (const float*)d_in[20];

  // workspace: [6*65536 bf16 weights][B*DEPTH*DF f32 zf history] (~51.1 MB total)
  __bf16* wbf = (__bf16*)d_ws;
  float*  hst = (float*)((char*)d_ws + (size_t)6 * 65536 * sizeof(__bf16));

  convert_weights_kernel<<<256, 256, 0, stream>>>(w_lin_r, w_lin_i, wv_r, wv_i,
                                                  wo_r, wo_i, wbf);
  crsn_kernel<<<Bz / MT, TPB, 0, stream>>>(z_real, z_imag, ptr_in,
                                           ln_scale, ln_shift, mod_bias,
                                           w_halt, b_halt, w_ctrl, b_ctrl,
                                           wbf, hst, (float*)d_out);
}
